// VectorQuantizer_20100446946154
// MI455X (gfx1250) — matrix operand order repaired and checked by reading, not compiled, not run on hardware
//
#include <hip/hip_runtime.h>
#include <hip/hip_bf16.h>
#include <math.h>

#define N_E   2048
#define E_DIM 256
#define N_TOK 65536
#define BETA  0.25f

typedef __attribute__((ext_vector_type(16))) __bf16 v16bf;
typedef __attribute__((ext_vector_type(8)))  float  v8f;

// gfx1250 async global->LDS copy (ASYNCcnt path). Diagnostic from the probe
// round shows the builtin takes (int4 addrspace(1)*, int4 addrspace(3)*, Ii, Ii).
#if __has_builtin(__builtin_amdgcn_global_load_async_to_lds_b128)
#define VQ_ASYNC_LDS 1
typedef int vq_v4i __attribute__((vector_size(16)));
typedef __attribute__((address_space(1))) vq_v4i* vq_gptr;
typedef __attribute__((address_space(3))) vq_v4i* vq_lptr;
#define VQ_ASYNC_CP16(gsrc, ldst) \
  __builtin_amdgcn_global_load_async_to_lds_b128((vq_gptr)(gsrc), (vq_lptr)(ldst), 0, 0)
#define VQ_WAIT_ASYNC() asm volatile("s_wait_asynccnt 0x0" ::: "memory")
#else
#define VQ_ASYNC_LDS 0
#endif

static __device__ inline v8f wmma_bf16f32(v16bf a, v16bf b, v8f c) {
  // 8 args: (neg_a, A, neg_b, B, c_mod, C, reuse_a, reuse_b)
  return __builtin_amdgcn_wmma_f32_16x16x32_bf16(false, a, false, b, (short)0, c,
                                                 false, false);
}

// ---------------------------------------------------------------------------
// Kernel 1: convert embedding to bf16 hi/lo split, compute ||e||^2, zero counts
// ---------------------------------------------------------------------------
__global__ void vq_prep(const float* __restrict__ E,
                        __bf16* __restrict__ ehi, __bf16* __restrict__ elo,
                        float* __restrict__ enorm, int* __restrict__ counts) {
  __shared__ float sred[8];
  const int r = blockIdx.x;
  const int c = threadIdx.x;
  float x = E[(size_t)r * E_DIM + c];
  __bf16 h = (__bf16)x;
  ehi[(size_t)r * E_DIM + c] = h;
  elo[(size_t)r * E_DIM + c] = (__bf16)(x - (float)h);
  float v = x * x;
  #pragma unroll
  for (int o = 16; o >= 1; o >>= 1) v += __shfl_xor(v, o, 32);
  if ((threadIdx.x & 31) == 0) sred[threadIdx.x >> 5] = v;
  __syncthreads();
  if (threadIdx.x == 0) {
    float s = 0.f;
    #pragma unroll
    for (int i = 0; i < 8; i++) s += sred[i];
    enorm[r] = s;
    counts[r] = 0;
  }
}

// ---------------------------------------------------------------------------
// Kernel 2: distance GEMM (bf16 split-3 WMMA, 3 independent accumulator
// chains) + per-row argmin. 8 waves/block, wave owns 16 token rows; B tiles
// double-buffered through LDS via async global->LDS loads.
// ---------------------------------------------------------------------------
__global__ __launch_bounds__(256) void vq_dist_argmin(
    const float* __restrict__ z,
    const __bf16* __restrict__ ehi, const __bf16* __restrict__ elo,
    const float* __restrict__ enorm,
    int* __restrict__ idx_i, float* __restrict__ out_idx,
    int* __restrict__ counts) {
  constexpr int NT = N_E / 16;  // 128 column tiles of 16 embeddings
  __shared__ __align__(16) __bf16 sB[2][2][4096];  // [buf][hi/lo] 32 KB
  __shared__ float sNorm[N_E];                      // 8 KB

  const int tid  = threadIdx.x;
  const int wave = tid >> 5;
  const int lane = tid & 31;
  const int nl   = lane & 15;
  const int kh   = lane >> 4;

  for (int i = tid; i < N_E; i += 256) sNorm[i] = enorm[i];

  // ---- load this wave's 16 z rows into bf16 hi/lo A fragments (K=256) ----
  const int r0 = blockIdx.x * 128 + wave * 16;
  const float* zr = z + (size_t)(r0 + nl) * E_DIM;
  v16bf ahi[8], alo[8];
  #pragma unroll
  for (int k = 0; k < 8; k++) {
    const int b1 = k * 32 + kh * 16;
    #pragma unroll
    for (int j = 0; j < 8; j++) {
      float x = zr[b1 + j];
      __bf16 h = (__bf16)x;
      ahi[k][j] = h;
      alo[k][j] = (__bf16)(x - (float)h);
      float y = zr[b1 + 8 + j];
      __bf16 h2 = (__bf16)y;
      ahi[k][8 + j] = h2;
      alo[k][8 + j] = (__bf16)(y - (float)h2);
    }
  }

  // ---- cooperative B-tile copy mapping: thread -> (kstep, lane-slot) ----
  const int ck    = tid >> 5;
  const int cl    = tid & 31;
  const int cn    = cl & 15;
  const int ckoff = ck * 32 + ((cl >> 4) << 4);
  const int cdst  = (ck * 32 + cl) * 16;

#if VQ_ASYNC_LDS
  {  // tile 0 via async copy
    size_t so = (size_t)cn * E_DIM + ckoff;
    VQ_ASYNC_CP16(&ehi[so],     &sB[0][0][cdst]);
    VQ_ASYNC_CP16(&ehi[so + 8], &sB[0][0][cdst + 8]);
    VQ_ASYNC_CP16(&elo[so],     &sB[0][1][cdst]);
    VQ_ASYNC_CP16(&elo[so + 8], &sB[0][1][cdst + 8]);
    VQ_WAIT_ASYNC();
  }
#else
  {  // tile 0 via regs
    size_t so = (size_t)cn * E_DIM + ckoff;
    const uint4* sh = (const uint4*)&ehi[so];
    const uint4* sl = (const uint4*)&elo[so];
    uint4* dh = (uint4*)&sB[0][0][cdst];
    uint4* dl = (uint4*)&sB[0][1][cdst];
    dh[0] = sh[0]; dh[1] = sh[1];
    dl[0] = sl[0]; dl[1] = sl[1];
  }
#endif
  __syncthreads();

  float minv[8];
  int   mini[8];
  #pragma unroll
  for (int i = 0; i < 8; i++) { minv[i] = __builtin_inff(); mini[i] = 0; }

  for (int t = 0; t < NT; t++) {
    const int cur = t & 1, nxt = cur ^ 1;
    const bool more = (t + 1 < NT);

#if VQ_ASYNC_LDS
    if (more) {  // async stage of next tile; overlaps with WMMAs below
      size_t so = (size_t)((t + 1) * 16 + cn) * E_DIM + ckoff;
      VQ_ASYNC_CP16(&ehi[so],     &sB[nxt][0][cdst]);
      VQ_ASYNC_CP16(&ehi[so + 8], &sB[nxt][0][cdst + 8]);
      VQ_ASYNC_CP16(&elo[so],     &sB[nxt][1][cdst]);
      VQ_ASYNC_CP16(&elo[so + 8], &sB[nxt][1][cdst + 8]);
    }
#else
    uint4 ph0, ph1, pl0, pl1;
    if (more) {
      size_t so = (size_t)((t + 1) * 16 + cn) * E_DIM + ckoff;
      const uint4* sh = (const uint4*)&ehi[so];
      const uint4* sl = (const uint4*)&elo[so];
      ph0 = sh[0]; ph1 = sh[1]; pl0 = sl[0]; pl1 = sl[1];
    }
#endif

    // three independent WMMA chains -> 3-way matrix-pipe ILP
    v8f acc0 = {0.f, 0.f, 0.f, 0.f, 0.f, 0.f, 0.f, 0.f};
    v8f acc1 = acc0, acc2 = acc0;
    #pragma unroll
    for (int k = 0; k < 8; k++) {
      const v16bf bh = *(const v16bf*)&sB[cur][0][(k * 32 + lane) * 16];
      const v16bf bl = *(const v16bf*)&sB[cur][1][(k * 32 + lane) * 16];
      acc0 = wmma_bf16f32(ahi[k], bh, acc0);  // hi*hi
      acc1 = wmma_bf16f32(alo[k], bh, acc1);  // lo*hi
      acc2 = wmma_bf16f32(ahi[k], bl, acc2);  // hi*lo
    }

    const int col = t * 16 + nl;
    const float en = sNorm[col];
    #pragma unroll
    for (int i = 0; i < 8; i++) {
      float d = en - 2.0f * ((acc0[i] + acc1[i]) + acc2[i]);
      if (d < minv[i]) { minv[i] = d; mini[i] = col; }
    }

    if (more) {
#if VQ_ASYNC_LDS
      VQ_WAIT_ASYNC();
#else
      uint4* dh = (uint4*)&sB[nxt][0][cdst];
      uint4* dl = (uint4*)&sB[nxt][1][cdst];
      dh[0] = ph0; dh[1] = ph1;
      dl[0] = pl0; dl[1] = pl1;
#endif
      __syncthreads();
    }
  }

  // cross-lane argmin within each 16-lane half (xor 8,4,2,1 stays in-half)
  #pragma unroll
  for (int off = 8; off >= 1; off >>= 1) {
    #pragma unroll
    for (int i = 0; i < 8; i++) {
      float ov = __shfl_xor(minv[i], off, 32);
      int   oi = __shfl_xor(mini[i], off, 32);
      if (ov < minv[i] || (ov == minv[i] && oi < mini[i])) {
        minv[i] = ov; mini[i] = oi;
      }
    }
  }

  if (nl == 0) {  // lanes 0 and 16 hold rows r0..r0+7 and r0+8..r0+15
    const int rb = r0 + kh * 8;
    #pragma unroll
    for (int i = 0; i < 8; i++) {
      const int r  = rb + i;
      const int id = mini[i];
      idx_i[r]   = id;
      out_idx[r] = (float)id;
      atomicAdd(&counts[id], 1);
    }
  }
}

// ---------------------------------------------------------------------------
// Kernel 3: gather z_q (fp32), write z_q_st (== z_q forward), per-row sq-err
// ---------------------------------------------------------------------------
__global__ void vq_gather_loss(const float* __restrict__ z,
                               const float* __restrict__ E,
                               const int* __restrict__ idx_i,
                               float* __restrict__ out_zq,
                               float* __restrict__ loss_row) {
  __shared__ float sred[8];
  const int r = blockIdx.x;
  const int c = threadIdx.x;
  const int id = idx_i[r];
  float q = E[(size_t)id * E_DIM + c];
  float x = z[(size_t)r * E_DIM + c];
  out_zq[(size_t)r * E_DIM + c] = q;
  float dz = q - x;
  float v = dz * dz;
  #pragma unroll
  for (int o = 16; o >= 1; o >>= 1) v += __shfl_xor(v, o, 32);
  if ((threadIdx.x & 31) == 0) sred[threadIdx.x >> 5] = v;
  __syncthreads();
  if (threadIdx.x == 0) {
    float s = 0.f;
    #pragma unroll
    for (int i = 0; i < 8; i++) s += sred[i];
    loss_row[r] = s;
  }
}

// ---------------------------------------------------------------------------
// Kernel 4: deterministic final reductions -> loss, perplexity
// ---------------------------------------------------------------------------
__global__ void vq_finalize(const float* __restrict__ loss_row,
                            const int* __restrict__ counts,
                            float* __restrict__ out_loss,
                            float* __restrict__ out_ppl) {
  __shared__ float sred[8];
  float s = 0.f;
  for (int i = threadIdx.x; i < N_TOK; i += 256) s += loss_row[i];
  #pragma unroll
  for (int o = 16; o >= 1; o >>= 1) s += __shfl_xor(s, o, 32);
  if ((threadIdx.x & 31) == 0) sred[threadIdx.x >> 5] = s;
  __syncthreads();
  if (threadIdx.x == 0) {
    float tot = 0.f;
    for (int i = 0; i < 8; i++) tot += sred[i];
    // beta*mean((sg(zq)-z)^2) + mean((zq-sg(z))^2) == (1+beta)*mean(diff^2)
    *out_loss = (1.0f + BETA) * tot / (float)((size_t)N_TOK * E_DIM);
  }
  __syncthreads();
  float h = 0.f;
  for (int i = threadIdx.x; i < N_E; i += 256) {
    float p = (float)counts[i] / (float)N_TOK;
    h -= p * logf(p + 1e-10f);
  }
  #pragma unroll
  for (int o = 16; o >= 1; o >>= 1) h += __shfl_xor(h, o, 32);
  if ((threadIdx.x & 31) == 0) sred[threadIdx.x >> 5] = h;
  __syncthreads();
  if (threadIdx.x == 0) {
    float tot = 0.f;
    for (int i = 0; i < 8; i++) tot += sred[i];
    *out_ppl = expf(tot);
  }
}

// ---------------------------------------------------------------------------
extern "C" void kernel_launch(void* const* d_in, const int* in_sizes, int n_in,
                              void* d_out, int out_size, void* d_ws, size_t ws_size,
                              hipStream_t stream) {
  const float* z = (const float*)d_in[0];  // (65536, 256) f32
  const float* E = (const float*)d_in[1];  // (2048, 256) f32

  // workspace layout (~2.6 MB)
  char* ws = (char*)d_ws;
  __bf16* ehi   = (__bf16*)ws;  ws += (size_t)N_E * E_DIM * sizeof(__bf16);
  __bf16* elo   = (__bf16*)ws;  ws += (size_t)N_E * E_DIM * sizeof(__bf16);
  float* enorm  = (float*)ws;   ws += (size_t)N_E * sizeof(float);
  int*   counts = (int*)ws;     ws += (size_t)N_E * sizeof(int);
  int*   idx_i  = (int*)ws;     ws += (size_t)N_TOK * sizeof(int);
  float* loss_row = (float*)ws; ws += (size_t)N_TOK * sizeof(float);

  // outputs flat in tuple order: z_q_st, loss, perplexity, idx
  float* out      = (float*)d_out;
  float* out_zq   = out;
  float* out_loss = out + (size_t)N_TOK * E_DIM;
  float* out_ppl  = out_loss + 1;
  float* out_idx  = out_ppl + 1;

  vq_prep<<<N_E, 256, 0, stream>>>(E, ehi, elo, enorm, counts);
  vq_dist_argmin<<<N_TOK / 128, 256, 0, stream>>>(z, ehi, elo, enorm,
                                                  idx_i, out_idx, counts);
  vq_gather_loss<<<N_TOK, 256, 0, stream>>>(z, E, idx_i, out_zq, loss_row);
  vq_finalize<<<1, 256, 0, stream>>>(loss_row, counts, out_loss, out_ppl);
}
